// MAMBA4Rec_53876069761562
// MI455X (gfx1250) — matrix operand order repaired and checked by reading, not compiled, not run on hardware
//
#include <hip/hip_runtime.h>
#include <hip/hip_bf16.h>

// ---------------------------------------------------------------------------
// MAMBA4Rec forward for MI455X (gfx1250, wave32, WMMA).
//  - GEMMs: v_wmma_f32_16x16x32_bf16, 128x128 block, wave tile 32x64 (2x4 frags)
//  - Selective scan: B/C bulk-preloaded to LDS via global_load_async_to_lds_b128
//    (ASYNCcnt path), barrier-free scan loop, fused SiLU gate.
// ---------------------------------------------------------------------------

#define CB   256      // batch
#define CL   200      // seq len
#define CT   300      // interleaved len (T = L + L/2)
#define CH   256      // hidden
#define CE   512      // expanded
#define CN   16       // state dim
#define CV   150000   // vocab
#define CBT  (CB * CT)

typedef __attribute__((ext_vector_type(8)))  unsigned short v8us;
typedef __attribute__((ext_vector_type(16))) unsigned short v16us;
typedef __attribute__((ext_vector_type(16))) __bf16         bf16x16;
typedef __attribute__((ext_vector_type(8)))  float          v8f;

#define BM 128
#define BN 128
#define BK 32
#define LDA_S 40   // LDS row stride (halfwords), padded, rows stay 16B aligned
#define LDB_S 40

__device__ __forceinline__ unsigned short f2bf(float f) {
    unsigned u = __float_as_uint(f);
    u += 0x7FFFu + ((u >> 16) & 1u);   // round-to-nearest-even
    return (unsigned short)(u >> 16);
}

__device__ __forceinline__ bf16x16 mk_frag(const unsigned short* p0,
                                           const unsigned short* p1) {
    v8us lo = *(const v8us*)p0;
    v8us hi = *(const v8us*)p1;
    v16us c = __builtin_shufflevector(lo, hi, 0, 1, 2, 3, 4, 5, 6, 7,
                                              8, 9, 10, 11, 12, 13, 14, 15);
    return __builtin_bit_cast(bf16x16, c);
}

// C[M,N] = A[M,K] @ W[N,K]^T ; mode: 0 = plain, 1 = +bias, 2 = +bias + GELU
__global__ __launch_bounds__(256) void gemm_bf16_wmma(
    const float* __restrict__ A, const float* __restrict__ W,
    const float* __restrict__ bias, float* __restrict__ C,
    int M, int N, int K, int mode)
{
    __shared__ unsigned short As[BM * LDA_S];
    __shared__ unsigned short Ws[BN * LDB_S];

    const int tid  = threadIdx.x;
    const int lane = tid & 31;
    const int wave = tid >> 5;
    const int wm   = wave & 3;   // wave row group: 32*wm
    const int wn   = wave >> 2;  // wave col group: 64*wn
    const int hlf  = lane >> 4;  // lane half (ISA K-swizzle select)
    const int lr   = lane & 15;

    const int blockN = blockIdx.x * BN;
    const int blockM = blockIdx.y * BM;

    v8f zero8;
#pragma unroll
    for (int q = 0; q < 8; ++q) zero8[q] = 0.0f;
    v8f acc[2][4];
#pragma unroll
    for (int i = 0; i < 2; ++i)
#pragma unroll
        for (int j = 0; j < 4; ++j) acc[i][j] = zero8;

    const int tr = tid >> 3;        // 0..31
    const int tc = (tid & 7) * 4;   // 0,4,...,28

    for (int k0 = 0; k0 < K; k0 += BK) {
        __syncthreads();
        // stage A tile (128x32) f32 -> bf16
#pragma unroll
        for (int p = 0; p < 4; ++p) {
            int row = tr + p * 32;
            int gm  = blockM + row;
            float4 v = make_float4(0.f, 0.f, 0.f, 0.f);
            if (gm < M) v = *(const float4*)(A + (size_t)gm * K + k0 + tc);
            unsigned short* d = &As[row * LDA_S + tc];
            d[0] = f2bf(v.x); d[1] = f2bf(v.y); d[2] = f2bf(v.z); d[3] = f2bf(v.w);
        }
        // stage W tile (128x32) f32 -> bf16
#pragma unroll
        for (int p = 0; p < 4; ++p) {
            int row = tr + p * 32;
            int gn  = blockN + row;
            float4 v = make_float4(0.f, 0.f, 0.f, 0.f);
            if (gn < N) v = *(const float4*)(W + (size_t)gn * K + k0 + tc);
            unsigned short* d = &Ws[row * LDB_S + tc];
            d[0] = f2bf(v.x); d[1] = f2bf(v.y); d[2] = f2bf(v.z); d[3] = f2bf(v.w);
        }
        __syncthreads();

        // A frag (16x32): lane half 0 -> K 0..7/16..23 ; half 1 -> K 8..15/24..31
        bf16x16 af[2], bfr[4];
#pragma unroll
        for (int i = 0; i < 2; ++i) {
            int row = wm * 32 + i * 16 + lr;
            const unsigned short* base = &As[row * LDA_S + hlf * 8];
            af[i] = mk_frag(base, base + 16);
        }
        // B frag (32x16): lane half 0 -> K 0..7/16..23 ; half 1 -> K 8..15/24..31
#pragma unroll
        for (int j = 0; j < 4; ++j) {
            int col = wn * 64 + j * 16 + lr;
            const unsigned short* base = &Ws[col * LDB_S + hlf * 8];
            bfr[j] = mk_frag(base, base + 16);
        }
#pragma unroll
        for (int i = 0; i < 2; ++i)
#pragma unroll
            for (int j = 0; j < 4; ++j)
                acc[i][j] = __builtin_amdgcn_wmma_f32_16x16x32_bf16(
                    false, af[i], false, bfr[j], (short)0, acc[i][j], false, false);
    }

    // C layout: VGPR v -> row (v + 8*half), col = lane%16
#pragma unroll
    for (int i = 0; i < 2; ++i) {
#pragma unroll
        for (int j = 0; j < 4; ++j) {
            int col = blockN + wn * 64 + j * 16 + lr;
            if (col >= N) continue;
            float bv = (mode >= 1) ? bias[col] : 0.0f;
#pragma unroll
            for (int v = 0; v < 8; ++v) {
                int row = blockM + wm * 32 + i * 16 + v + hlf * 8;
                if (row >= M) continue;
                float val = acc[i][j][v] + bv;
                if (mode == 2)
                    val = 0.5f * val * (1.0f + erff(val * 0.70710678118654752f));
                C[(size_t)row * N + col] = val;
            }
        }
    }
}

// ---------------- embedding gather with interleave ------------------------
// t = 3m -> seq[2m] ; t = 3m+1 -> seq[2m+1] ; t = 3m+2 -> V-1 (user token)
__global__ void embed_kernel(const int* __restrict__ seq,
                             const float* __restrict__ emb,
                             float* __restrict__ out)
{
    size_t gid = (size_t)blockIdx.x * 256 + threadIdx.x;   // over B*T*H
    int f = (int)(gid & (CH - 1));
    size_t bt = gid >> 8;
    int b = (int)(bt / CT);
    int t = (int)(bt - (size_t)b * CT);
    int m = t / 3, r = t - 3 * m;
    int idx = (r == 2) ? (CV - 1) : seq[b * CL + 2 * m + r];
    out[gid] = emb[(size_t)idx * CH + f];
}

// ---------------- LayerNorm (in-place safe; optional residual) ------------
__global__ __launch_bounds__(256) void ln_kernel(
    const float* __restrict__ x, const float* __restrict__ res,
    const float* __restrict__ w, const float* __restrict__ b,
    float* __restrict__ out)
{
    __shared__ float rs[256];
    __shared__ float rq[256];
    int row = blockIdx.x;
    int t = threadIdx.x;
    size_t idx = (size_t)row * CH + t;
    float v = x[idx];
    if (res) v += res[idx];
    rs[t] = v; rq[t] = v * v;
    __syncthreads();
    for (int s = 128; s > 0; s >>= 1) {
        if (t < s) { rs[t] += rs[t + s]; rq[t] += rq[t + s]; }
        __syncthreads();
    }
    float mean = rs[0] * (1.0f / CH);
    float var  = rq[0] * (1.0f / CH) - mean * mean;
    float inv  = rsqrtf(var + 1e-12f);
    out[idx] = (v - mean) * inv * w[t] + b[t];
}

// ---------------- depthwise causal conv (K=4) + SiLU ----------------------
__global__ void conv_silu_kernel(const float* __restrict__ xz,
                                 const float* __restrict__ cw,
                                 const float* __restrict__ cb,
                                 float* __restrict__ xc)
{
    size_t gid = (size_t)blockIdx.x * 256 + threadIdx.x;   // over B*T*E
    int e = (int)(gid & (CE - 1));
    size_t bt = gid >> 9;
    int b = (int)(bt / CT);
    int t = (int)(bt - (size_t)b * CT);
    float acc = cb[e];
#pragma unroll
    for (int k = 0; k < 4; ++k) {
        int ts = t - 3 + k;
        if (ts >= 0)
            acc += cw[e * 4 + k] * xz[((size_t)b * CT + ts) * (2 * CE) + e];
    }
    xc[gid] = acc / (1.0f + __expf(-acc));
}

// ---------------- dt = softplus(xdbl[:, :16] @ dt_w^T + dt_b) -------------
__global__ void dt_kernel(const float* __restrict__ xdbl,
                          const float* __restrict__ dtw,
                          const float* __restrict__ dtb,
                          float* __restrict__ dtv)
{
    size_t gid = (size_t)blockIdx.x * 256 + threadIdx.x;   // over B*T*E
    int e = (int)(gid & (CE - 1));
    size_t bt = gid >> 9;
    const float* xr = xdbl + bt * 48;
    float acc = dtb[e];
#pragma unroll
    for (int r = 0; r < 16; ++r) acc += xr[r] * dtw[e * 16 + r];
    dtv[gid] = (acc > 20.0f) ? acc : log1pf(__expf(acc));
}

// ---------------- selective scan + fused SiLU gate ------------------------
// One block per batch, one thread per E channel. All 300 steps of B/C
// (300 x 32 floats = 38.4 KB) are bulk-copied to LDS with the gfx1250
// async-to-LDS path, then the scan loop runs with zero barriers and
// software prefetch of next-step x/dt/z.
__global__ __launch_bounds__(512) void scan_kernel(
    const float* __restrict__ xc, const float* __restrict__ dtv,
    const float* __restrict__ xdbl, const float* __restrict__ xz,
    const float* __restrict__ A_log, const float* __restrict__ Dp,
    float* __restrict__ y)
{
    __shared__ float bc[CT * 32];   // [t][0:16]=B, [t][16:32]=C
    const int b = blockIdx.x;
    const int e = threadIdx.x;

    // async bulk copy: xdbl[b, t, 16:48] -> bc[t][0:32], 16B per lane-op
    {
        const float* src = xdbl + (size_t)b * CT * 48;
        for (int i = threadIdx.x; i < CT * 8; i += 512) {
            int t = i >> 3, c = i & 7;
            unsigned lds = (unsigned)(unsigned long long)&bc[t * 32 + c * 4];
            unsigned long long ga =
                (unsigned long long)(src + (size_t)t * 48 + 16 + c * 4);
            asm volatile("global_load_async_to_lds_b128 %0, %1, off"
                         :: "v"(lds), "v"(ga) : "memory");
        }
        asm volatile("s_wait_asynccnt 0x0" ::: "memory");
    }
    __syncthreads();

    float Aa[CN];
#pragma unroll
    for (int n = 0; n < CN; ++n) Aa[n] = -__expf(A_log[e * CN + n]);
    float Dv = Dp[e];
    float s[CN];
#pragma unroll
    for (int n = 0; n < CN; ++n) s[n] = 0.0f;

    const size_t base  = (size_t)b * CT * CE + e;
    const size_t zbase = (size_t)b * CT * (2 * CE) + CE + e;

    float x = xc[base];
    float d = dtv[base];
    float z = xz[zbase];

    for (int t = 0; t < CT; ++t) {
        float xn = 0.f, dn = 0.f, zn = 0.f;
        if (t + 1 < CT) {                      // prefetch next step
            xn = xc[base + (size_t)(t + 1) * CE];
            dn = dtv[base + (size_t)(t + 1) * CE];
            zn = xz[zbase + (size_t)(t + 1) * (2 * CE)];
        }
        const float* bcp = &bc[t * 32];
        float dx = d * x;
        float accy = 0.0f;
#pragma unroll
        for (int n = 0; n < CN; ++n) {
            s[n] = s[n] * __expf(d * Aa[n]) + dx * bcp[n];
            accy += s[n] * bcp[16 + n];
        }
        float sz = z / (1.0f + __expf(-z));    // fused gate
        y[base + (size_t)t * CE] = (accy + x * Dv) * sz;
        x = xn; d = dn; z = zn;
    }
}

// ---------------- final gather at clip(seq_len + seq_len/2 - 1) -----------
__global__ void gather_kernel(const int* __restrict__ seq_len,
                              const float* __restrict__ h,
                              float* __restrict__ out)
{
    int gid = blockIdx.x * 256 + threadIdx.x;   // B*H = 65536
    int b = gid >> 8;
    int f = gid & (CH - 1);
    int sl = seq_len[b];
    int idx = sl + sl / 2 - 1;
    idx = idx < 0 ? 0 : (idx > CT - 1 ? CT - 1 : idx);
    out[gid] = h[((size_t)b * CT + idx) * CH + f];
}

// ---------------------------------------------------------------------------
extern "C" void kernel_launch(void* const* d_in, const int* in_sizes, int n_in,
                              void* d_out, int out_size, void* d_ws, size_t ws_size,
                              hipStream_t stream)
{
    (void)in_sizes; (void)n_in; (void)out_size; (void)ws_size;

    const int*   item_seq  = (const int*)  d_in[0];
    const int*   seq_len   = (const int*)  d_in[1];
    const float* item_emb  = (const float*)d_in[2];
    const float* ln_w      = (const float*)d_in[3];
    const float* ln_b      = (const float*)d_in[4];
    const float* in_proj_w = (const float*)d_in[5];
    const float* conv_w    = (const float*)d_in[6];
    const float* conv_b    = (const float*)d_in[7];
    const float* x_proj_w  = (const float*)d_in[8];
    const float* dt_w      = (const float*)d_in[9];
    const float* dt_b      = (const float*)d_in[10];
    const float* A_log     = (const float*)d_in[11];
    const float* Dp        = (const float*)d_in[12];
    const float* out_w     = (const float*)d_in[13];
    const float* mn_w      = (const float*)d_in[14];
    const float* mn_b      = (const float*)d_in[15];
    const float* fc1_w     = (const float*)d_in[16];
    const float* fc1_b     = (const float*)d_in[17];
    const float* fc2_w     = (const float*)d_in[18];
    const float* fc2_b     = (const float*)d_in[19];
    const float* fn_w      = (const float*)d_in[20];
    const float* fn_b      = (const float*)d_in[21];

    // workspace layout (bytes, all 256-aligned)
    char* ws = (char*)d_ws;
    float* h    = (float*)(ws + 0);             // BT*256
    float* xz   = (float*)(ws + 78643200u);     // BT*1024 (also ff1)
    float* xc   = (float*)(ws + 393216000u);    // BT*512
    float* xdbl = (float*)(ws + 550502400u);    // BT*48
    float* dtv  = (float*)(ws + 565248000u);    // BT*512
    float* y    = (float*)(ws + 722534400u);    // BT*512 (gated in scan)
    float* mo   = (float*)(ws + 879820800u);    // BT*256 (also ff2)

    const int M = CBT;                   // 76800 rows
    const int gBTH = CBT * CH / 256;     // 76800
    const int gBTE = CBT * CE / 256;     // 153600

    // embedding + LN
    embed_kernel<<<gBTH, 256, 0, stream>>>(item_seq, item_emb, h);
    ln_kernel<<<CBT, 256, 0, stream>>>(h, nullptr, ln_w, ln_b, h);

    for (int l = 0; l < 2; ++l) {
        const float* ipw = in_proj_w + (size_t)l * (2 * CE) * CH;
        const float* cwl = conv_w + (size_t)l * CE * 4;
        const float* cbl = conv_b + (size_t)l * CE;
        const float* xpw = x_proj_w + (size_t)l * 48 * CE;
        const float* dwl = dt_w + (size_t)l * CE * 16;
        const float* dbl_ = dt_b + (size_t)l * CE;
        const float* alw = A_log + (size_t)l * CE * CN;
        const float* dpl = Dp + (size_t)l * CE;
        const float* opw = out_w + (size_t)l * CH * CE;
        const float* f1w = fc1_w + (size_t)l * (4 * CH) * CH;
        const float* f1b = fc1_b + (size_t)l * (4 * CH);
        const float* f2w = fc2_w + (size_t)l * CH * (4 * CH);
        const float* f2b = fc2_b + (size_t)l * CH;

        // xz = h @ in_proj_w^T   (76800 x 1024, K=256)
        gemm_bf16_wmma<<<dim3((2 * CE) / BN, M / BM), 256, 0, stream>>>(
            h, ipw, nullptr, xz, M, 2 * CE, CH, 0);
        // xc = silu(causal depthwise conv + b)
        conv_silu_kernel<<<gBTE, 256, 0, stream>>>(xz, cwl, cbl, xc);
        // xdbl = xc @ x_proj_w^T (76800 x 48, K=512)
        gemm_bf16_wmma<<<dim3(1, M / BM), 256, 0, stream>>>(
            xc, xpw, nullptr, xdbl, M, 48, CE, 0);
        // dt = softplus(xdbl[:, :16] @ dt_w^T + dt_b)
        dt_kernel<<<gBTE, 256, 0, stream>>>(xdbl, dwl, dbl_, dtv);
        // selective scan (async LDS preload, fused gate)
        scan_kernel<<<CB, CE, 0, stream>>>(xc, dtv, xdbl, xz, alw, dpl, y);
        // mo = y @ out_w^T (76800 x 256, K=512)
        gemm_bf16_wmma<<<dim3((CH + BN - 1) / BN, M / BM), 256, 0, stream>>>(
            y, opw, nullptr, mo, M, CH, CE, 0);
        // h = LN(mo + h)
        ln_kernel<<<CBT, 256, 0, stream>>>(mo, h, mn_w + l * CH, mn_b + l * CH, h);
        // ff = gelu(h @ fc1^T + b)  (76800 x 1024, K=256), fused epilogue
        gemm_bf16_wmma<<<dim3((4 * CH) / BN, M / BM), 256, 0, stream>>>(
            h, f1w, f1b, xz, M, 4 * CH, CH, 2);
        // ff2 = ff @ fc2^T + b (76800 x 256, K=1024)
        gemm_bf16_wmma<<<dim3((CH + BN - 1) / BN, M / BM), 256, 0, stream>>>(
            xz, f2w, f2b, mo, M, CH, 4 * CH, 1);
        // h = LN(ff2 + h)
        ln_kernel<<<CBT, 256, 0, stream>>>(mo, h, fn_w + l * CH, fn_b + l * CH, h);
    }

    // output: h[b, clip(tl-1)]  -> (256, 256) f32
    gather_kernel<<<CB * CH / 256, 256, 0, stream>>>(seq_len, h, (float*)d_out);
}
